// MultiHeadAttention_89404039234060
// MI455X (gfx1250) — matrix operand order repaired and checked by reading, not compiled, not run on hardware
//
#include <hip/hip_runtime.h>
#include <hip/hip_bf16.h>

// MI455X / gfx1250: wave32, WMMA 16x16x32 bf16 (f32 accum), async global->LDS,
// Tensor Data Mover for attention K/V tiles (padded-LDS 2D tile DMA).
// Pipeline: f32->bf16 convert + mask bit-pack | QKV GEMMs (double-buffered async
// LDS tiles, WMMA, split-head epilogue) | flash attention (64-key tiles, online
// softmax, packed-bit mask, TDM K/V tiles) | output GEMM (f32 -> d_out).

typedef __attribute__((ext_vector_type(16))) __bf16 v16bf;
typedef __attribute__((ext_vector_type(8)))  float  v8f;

#define B_  2
#define S_  2048
#define D_  1024
#define H_  16
#define DK_ 64
#define M_  (B_ * S_)   // 4096
#define SW_ (S_ / 32)   // mask words per row

// ---- gfx1250 async global->LDS (ASYNCcnt) with sync fallback --------------
#if defined(__has_builtin)
#  if __has_builtin(__builtin_amdgcn_global_load_async_to_lds_b128) && \
      __has_builtin(__builtin_amdgcn_s_wait_asynccnt)
#    define HAVE_ASYNC 1
#  endif
#  if __has_builtin(__builtin_amdgcn_tensor_load_to_lds) && \
      __has_builtin(__builtin_amdgcn_s_wait_tensorcnt)
#    define HAVE_TDM 1
#  endif
#endif
#ifndef HAVE_ASYNC
#  define HAVE_ASYNC 0
#endif
#ifndef HAVE_TDM
#  define HAVE_TDM 0
#endif

#if HAVE_ASYNC
typedef int avi4 __attribute__((vector_size(16)));  // builtin's b128 pointee type
#define ASG(p) ((__attribute__((address_space(1))) avi4*)(unsigned long long)(p))
#define ASL(p) ((__attribute__((address_space(3))) avi4*)(unsigned)(unsigned long long)(p))
#define ASYNC_CP16(g, l) __builtin_amdgcn_global_load_async_to_lds_b128(ASG(g), ASL(l), 0, 0)
#define ASYNC_WAIT(n)    __builtin_amdgcn_s_wait_asynccnt(n)
#else
#define ASYNC_CP16(g, l) (*(uint4*)(l) = *(const uint4*)(g))
#define ASYNC_WAIT(n)    ((void)0)
#endif

#if HAVE_TDM
typedef unsigned tdm_u4 __attribute__((ext_vector_type(4)));
typedef int      tdm_i8 __attribute__((ext_vector_type(8)));
typedef int      tdm_i4 __attribute__((ext_vector_type(4)));
// 2D tile DMA: 64 rows x 64 bf16 (128B) per row, row stride DK_ elems in memory,
// LDS rows padded by 16B every 128B (pad_interval code 4 = 32 DW, pad_amount
// code 3 = 4 DW) -> 144B LDS row stride, matching the [64][72] layout.
__device__ __forceinline__ void tdm_load_2d(const void* gsrc, void* ldst) {
    unsigned long long ga = (unsigned long long)gsrc;
    tdm_u4 g0;
    g0.x = 1u;                                    // count=1, user descriptor
    g0.y = (unsigned)(unsigned long long)ldst;    // lds_addr
    g0.z = (unsigned)ga;                          // global_addr[31:0]
    g0.w = (unsigned)(ga >> 32) | (2u << 30);     // global_addr[56:32] | type=2
    tdm_i8 g1;
    g1[0] = (1 << 16) | (1 << 20) | (4 << 22) | (3 << 25); // 2B elems, pad en
    g1[1] = DK_ << 16;          // tensor_dim0 = 64 (low bits in [63:48])
    g1[2] = S_ << 16;           // tensor_dim1 low bits in [95:80]
    g1[3] = 64 << 16;           // tile_dim0 = 64 ([127:112])
    g1[4] = 64;                 // tile_dim1 = 64 ([143:128]), tile_dim2 = 0
    g1[5] = DK_;                // tensor_dim0_stride = 64 ([191:160])
    g1[6] = 0;
    g1[7] = 0;
    tdm_i4 z4 = {0, 0, 0, 0};
#if __clang_major__ >= 23
    tdm_i8 z8 = {0, 0, 0, 0, 0, 0, 0, 0};
    __builtin_amdgcn_tensor_load_to_lds(g0, g1, z4, z4, z8, 0);
#else
    __builtin_amdgcn_tensor_load_to_lds(g0, g1, z4, z4, 0);
#endif
}
#define TDM_WAIT(n) __builtin_amdgcn_s_wait_tensorcnt(n)
#endif

__device__ __forceinline__ __bf16 f2bf(float f) {
    union { float f; unsigned u; } x; x.f = f;
    unsigned r = (x.u + 0x7FFFu + ((x.u >> 16) & 1u)) >> 16;
    union { unsigned short s; __bf16 b; } y; y.s = (unsigned short)r;
    return y.b;
}

// 1-VALU bf16 store: round then store high half (lowers to *_store_d16_hi_b16)
__device__ __forceinline__ void bstore(__bf16* p, float f) {
    unsigned u = __float_as_uint(f) + 0x8000u;
    *(unsigned short*)p = (unsigned short)(u >> 16);
}

// ---------------------------------------------------------------- convert ---
__global__ void cvt_f32_bf16(const float* __restrict__ s, __bf16* __restrict__ d, int n) {
    int i = (blockIdx.x * blockDim.x + threadIdx.x) * 4;
    int stride = gridDim.x * blockDim.x * 4;
    for (; i < n; i += stride) {
        float4 f = *(const float4*)(s + i);
        unsigned a = __float_as_uint(f.x) + 0x8000u;
        unsigned b = __float_as_uint(f.y) + 0x8000u;
        unsigned c = __float_as_uint(f.z) + 0x8000u;
        unsigned w = __float_as_uint(f.w) + 0x8000u;
        uint2 o;                      // one v_perm_b32 packs two bf16
        o.x = __builtin_amdgcn_perm(b, a, 0x07060302u);
        o.y = __builtin_amdgcn_perm(w, c, 0x07060302u);
        *(uint2*)(d + i) = o;
    }
}

// bool bytes -> packed u32 bit words (bit k of word = mask[word*32+k])
__global__ void pack_mask(const unsigned char* __restrict__ m,
                          unsigned* __restrict__ out, int nwords) {
    int i = blockIdx.x * blockDim.x + threadIdx.x;
    if (i >= nwords) return;
    const uint4* p = (const uint4*)(m + (size_t)i * 32);
    uint4 a = p[0], b = p[1];
    unsigned w = 0;
#pragma unroll
    for (int j = 0; j < 4; j++) {
        w |= ((a.x >> (8 * j)) & 1u) << (0 + j);
        w |= ((a.y >> (8 * j)) & 1u) << (4 + j);
        w |= ((a.z >> (8 * j)) & 1u) << (8 + j);
        w |= ((a.w >> (8 * j)) & 1u) << (12 + j);
        w |= ((b.x >> (8 * j)) & 1u) << (16 + j);
        w |= ((b.y >> (8 * j)) & 1u) << (20 + j);
        w |= ((b.z >> (8 * j)) & 1u) << (24 + j);
        w |= ((b.w >> (8 * j)) & 1u) << (28 + j);
    }
    out[i] = w;
}

// ------------------------------------------------------------------- GEMM ---
// C[M,N] = A[M,K] @ W[K,N] + bias.  M=4096, N=K=1024.
// Double-buffered async LDS tiles (BK=32).
// EPI=0: bf16 out, split-head [B,H,S,DK].  EPI=1: f32 out row-major (d_out).
template <int EPI>
__global__ __launch_bounds__(256) void gemm_bf16(
    const __bf16* __restrict__ A, const __bf16* __restrict__ W,
    const float* __restrict__ bias, void* __restrict__ outp) {
    const int K = D_, N = D_;
    __shared__ __bf16 As[2][128][32];
    __shared__ __bf16 Ws[2][32][136];   // row stride 272B (16B aligned)

    const int tid  = threadIdx.x;
    const int wave = tid >> 5;
    const int lane = tid & 31;
    const int half = lane >> 4;
    const int l15  = lane & 15;
    const int bm = blockIdx.y * 128;
    const int bn = blockIdx.x * 128;
    const int wm = (wave & 3) * 32;   // 4 waves along M
    const int wn = (wave >> 2) * 64;  // 2 waves along N

    v8f acc[2][4] = {};

    const int rowA = tid >> 1, cA = (tid & 1) * 16;
    const int rowW = tid >> 3, cW = (tid & 7) * 16;
    auto issue = [&](int k0, int buf) {  // 4 async instrs per wave
        const __bf16* gA = A + (size_t)(bm + rowA) * K + k0 + cA;
        const __bf16* gW = W + (size_t)(k0 + rowW) * N + bn + cW;
        __bf16* lA = &As[buf][rowA][cA];
        __bf16* lW = &Ws[buf][rowW][cW];
        ASYNC_CP16(gA,     lA);
        ASYNC_CP16(gA + 8, lA + 8);
        ASYNC_CP16(gW,     lW);
        ASYNC_CP16(gW + 8, lW + 8);
    };

    issue(0, 0);
    const int NT = K / 32;
    for (int i = 0; i < NT; i++) {
        const int buf = i & 1;
        __syncthreads();                       // prev readers of buf^1 done
        if (i + 1 < NT) {
            issue((i + 1) * 32, buf ^ 1);      // overlap next tile with compute
            ASYNC_WAIT(4);                     // current tile complete
        } else {
            ASYNC_WAIT(0);
        }
        __syncthreads();                       // all waves' tile data visible

        v16bf af[2];
#pragma unroll
        for (int ii = 0; ii < 2; ii++) {
            int row = wm + ii * 16 + l15;
#pragma unroll
            for (int v = 0; v < 8; v++) {
                int kk = 2 * v + (v >= 4 ? 8 : 0) + half * 8;
                af[ii][2 * v]     = As[buf][row][kk];
                af[ii][2 * v + 1] = As[buf][row][kk + 1];
            }
        }
        v16bf bfv[4];
#pragma unroll
        for (int j = 0; j < 4; j++) {
            int col = wn + j * 16 + l15;
#pragma unroll
            for (int v = 0; v < 8; v++) {
                int kk = 2 * v + (v >= 4 ? 8 : 0) + half * 8;
                bfv[j][2 * v]     = Ws[buf][kk][col];
                bfv[j][2 * v + 1] = Ws[buf][kk + 1][col];
            }
        }
#pragma unroll
        for (int ii = 0; ii < 2; ii++)
#pragma unroll
            for (int j = 0; j < 4; j++)
                acc[ii][j] = __builtin_amdgcn_wmma_f32_16x16x32_bf16(
                    false, af[ii], false, bfv[j], (short)0, acc[ii][j], false, false);
    }

#pragma unroll
    for (int ii = 0; ii < 2; ii++)
#pragma unroll
        for (int j = 0; j < 4; j++) {
            int n = bn + wn + j * 16 + l15;
            float bv = bias[n];
#pragma unroll
            for (int v = 0; v < 8; v++) {
                int m = bm + wm + ii * 16 + v + half * 8;
                float val = acc[ii][j][v] + bv;
                if (EPI == 0) {  // bf16, split-head [B,H,S,DK]
                    int b = m >> 11, s = m & (S_ - 1);
                    int h = n >> 6,  d = n & 63;
                    bstore((__bf16*)outp + (((size_t)(b * H_ + h)) * S_ + s) * DK_ + d, val);
                } else {         // f32, [M,N]
                    ((float*)outp)[(size_t)m * N + n] = val;
                }
            }
        }
}

// -------------------------------------------------------- flash attention ---
// grid (S/64, H, B); 4 waves/block, each wave owns 16 query rows.
// 64-key tiles, double-buffered TDM (or async) K/V, packed bitmask, online
// softmax.
__global__ __launch_bounds__(128) void attn(
    const __bf16* __restrict__ Qh, const __bf16* __restrict__ Kh,
    const __bf16* __restrict__ Vh, const unsigned* __restrict__ mbits,
    __bf16* __restrict__ O) {
    const int b = blockIdx.z, h = blockIdx.y;
    const int qbase = blockIdx.x * 64;
    const int wave = threadIdx.x >> 5, lane = threadIdx.x & 31;
    const int half = lane >> 4, l15 = lane & 15;
    const int qrow0 = qbase + wave * 16;

    const __bf16* Qp = Qh + ((size_t)(b * H_ + h)) * S_ * DK_;
    const __bf16* Kp = Kh + ((size_t)(b * H_ + h)) * S_ * DK_;
    const __bf16* Vp = Vh + ((size_t)(b * H_ + h)) * S_ * DK_;
    const unsigned* mb = mbits + (size_t)b * S_ * SW_;

    __shared__ __bf16 Ks[2][64][72];   // [key][d], 144B rows (TDM pad fields)
    __shared__ __bf16 Vs[2][64][72];   // [key][d], 144B rows
    __shared__ __bf16 Ps[4][16][72];   // per-wave P re-layout scratch

    // Q fragments (16x64 = 2 A-frags of K=32)
    v16bf qf[2];
#pragma unroll
    for (int f = 0; f < 2; f++) {
        int row = qrow0 + l15;
#pragma unroll
        for (int v = 0; v < 8; v++) {
            int kk = f * 32 + 2 * v + (v >= 4 ? 8 : 0) + half * 8;
            qf[f][2 * v]     = Qp[(size_t)row * DK_ + kk];
            qf[f][2 * v + 1] = Qp[(size_t)row * DK_ + kk + 1];
        }
    }

    v8f accO[4] = {};
    float m_run[8], l_run[8];
#pragma unroll
    for (int v = 0; v < 8; v++) { m_run[v] = -1e30f; l_run[v] = 0.0f; }
    const float scale = 0.125f;  // 1/sqrt(64)

#if HAVE_TDM
    auto issueTile = [&](int kb, int buf) {   // one wave drives the TDM
        if (wave == 0) {
            tdm_load_2d(Kp + (size_t)kb * DK_, &Ks[buf][0][0]);
            tdm_load_2d(Vp + (size_t)kb * DK_, &Vs[buf][0][0]);
        }
    };
#else
    const int krow = threadIdx.x >> 1;        // 64 rows, 2 threads/row
    const int kcol = (threadIdx.x & 1) * 32;  // 32 elems = 64B = 4 x b128
    auto issueTile = [&](int kb, int buf) {   // 8 async instrs per wave
        const __bf16* gK = Kp + (size_t)(kb + krow) * DK_ + kcol;
        const __bf16* gV = Vp + (size_t)(kb + krow) * DK_ + kcol;
        __bf16* lK = &Ks[buf][krow][kcol];
        __bf16* lV = &Vs[buf][krow][kcol];
#pragma unroll
        for (int q = 0; q < 4; q++) {
            ASYNC_CP16(gK + 8 * q, lK + 8 * q);
            ASYNC_CP16(gV + 8 * q, lV + 8 * q);
        }
    };
#endif

    issueTile(0, 0);
    const int NT = S_ / 64;
    for (int i = 0; i < NT; i++) {
        const int kb = i * 64, buf = i & 1;
        __syncthreads();                       // prev readers of buf^1 done
        if (i + 1 < NT) {
            issueTile(kb + 64, buf ^ 1);       // overlap next K/V with compute
#if HAVE_TDM
            if (wave == 0) TDM_WAIT(2);        // current tile complete
#else
            ASYNC_WAIT(8);
#endif
        } else {
#if HAVE_TDM
            if (wave == 0) TDM_WAIT(0);
#else
            ASYNC_WAIT(0);
#endif
        }
        __syncthreads();

        // scores S = Q K^T : four 16x16 col-tiles, contraction over d (2x32)
        v8f sc[4];
#pragma unroll
        for (int t = 0; t < 4; t++) {
            v8f c = {};
#pragma unroll
            for (int f = 0; f < 2; f++) {
                v16bf bfrag;
#pragma unroll
                for (int v = 0; v < 8; v++) {
                    int dd = f * 32 + 2 * v + (v >= 4 ? 8 : 0) + half * 8;  // d index
                    int key = t * 16 + l15;
                    bfrag[2 * v]     = Ks[buf][key][dd];  // B[d][key] = K[key][d]
                    bfrag[2 * v + 1] = Ks[buf][key][dd + 1];
                }
                c = __builtin_amdgcn_wmma_f32_16x16x32_bf16(
                    false, qf[f], false, bfrag, (short)0, c, false, false);
            }
            sc[t] = c;
        }

        // scale + packed-bit mask (C/D layout: row = v + half*8, col = lane&15)
        float p[4][8];
#pragma unroll
        for (int v = 0; v < 8; v++) {
            int q = qrow0 + v + half * 8;
            uint2 w = *(const uint2*)(mb + (size_t)q * SW_ + (kb >> 5));
#pragma unroll
            for (int t = 0; t < 4; t++) {
                unsigned word = (t < 2) ? w.x : w.y;
                int bit = (t & 1) * 16 + l15;
                float s = sc[t][v] * scale;
                p[t][v] = ((word >> bit) & 1u) ? s : -1e9f;
            }
        }

        // online softmax (width-16 xor reductions match lane-half row groups)
#pragma unroll
        for (int v = 0; v < 8; v++) {
            float mx = fmaxf(fmaxf(p[0][v], p[1][v]), fmaxf(p[2][v], p[3][v]));
#pragma unroll
            for (int off = 8; off; off >>= 1) mx = fmaxf(mx, __shfl_xor(mx, off, 16));
            float mnew = fmaxf(m_run[v], mx);
            float corr = __expf(m_run[v] - mnew);
            m_run[v] = mnew;
            float rs = 0.0f;
#pragma unroll
            for (int t = 0; t < 4; t++) { p[t][v] = __expf(p[t][v] - mnew); rs += p[t][v]; }
#pragma unroll
            for (int off = 8; off; off >>= 1) rs += __shfl_xor(rs, off, 16);
            l_run[v] = l_run[v] * corr + rs;
#pragma unroll
            for (int j = 0; j < 4; j++) accO[j][v] *= corr;
        }

        // C/D layout -> A layout via per-wave LDS scratch (d16_hi b16 stores)
#pragma unroll
        for (int t = 0; t < 4; t++)
#pragma unroll
            for (int v = 0; v < 8; v++)
                bstore(&Ps[wave][v + half * 8][t * 16 + l15], p[t][v]);
        asm volatile("" ::: "memory");  // DS ops are in-order per wave

        v16bf pf[2];
#pragma unroll
        for (int f = 0; f < 2; f++)
#pragma unroll
            for (int v = 0; v < 8; v++) {
                int kk = f * 32 + 2 * v + (v >= 4 ? 8 : 0) + half * 8;  // key
                pf[f][2 * v]     = Ps[wave][l15][kk];
                pf[f][2 * v + 1] = Ps[wave][l15][kk + 1];
            }

        // O += P V   (K = 64 keys = 2x32; 4 dim-tiles of 16)
#pragma unroll
        for (int f = 0; f < 2; f++)
#pragma unroll
            for (int j = 0; j < 4; j++) {
                v16bf vfrag;
#pragma unroll
                for (int v = 0; v < 8; v++) {
                    int kk = f * 32 + 2 * v + (v >= 4 ? 8 : 0) + half * 8;  // key index
                    int d = j * 16 + l15;
                    vfrag[2 * v]     = Vs[buf][kk][d];    // B[key][d]
                    vfrag[2 * v + 1] = Vs[buf][kk + 1][d];
                }
                accO[j] = __builtin_amdgcn_wmma_f32_16x16x32_bf16(
                    false, pf[f], false, vfrag, (short)0, accO[j], false, false);
            }
    }

    // normalize and write O in merged-head [B,S,D] layout (bf16)
#pragma unroll
    for (int v = 0; v < 8; v++) {
        float inv = 1.0f / l_run[v];
        int srow = qrow0 + v + half * 8;
#pragma unroll
        for (int j = 0; j < 4; j++) {
            int d = j * 16 + l15;
            bstore(O + ((size_t)b * S_ + srow) * D_ + h * DK_ + d, accO[j][v] * inv);
        }
    }
}

// ---------------------------------------------------------------- launch ---
extern "C" void kernel_launch(void* const* d_in, const int* in_sizes, int n_in,
                              void* d_out, int out_size, void* d_ws, size_t ws_size,
                              hipStream_t stream) {
    const float* q  = (const float*)d_in[0];
    const float* k  = (const float*)d_in[1];
    const float* v  = (const float*)d_in[2];
    const float* Wq = (const float*)d_in[3];
    const float* bq = (const float*)d_in[4];
    const float* Wk = (const float*)d_in[5];
    const float* bk = (const float*)d_in[6];
    const float* Wv = (const float*)d_in[7];
    const float* bv = (const float*)d_in[8];
    const float* Wo = (const float*)d_in[9];
    const float* bo = (const float*)d_in[10];
    const unsigned char* mask = (const unsigned char*)d_in[11];

    char* ws = (char*)d_ws;
    size_t off = 0;
    const size_t XN = (size_t)M_ * D_;   // 4M elems
    const size_t WN = (size_t)D_ * D_;   // 1M elems
    __bf16* Xq  = (__bf16*)(ws + off); off += XN * 2;
    __bf16* Xk  = (__bf16*)(ws + off); off += XN * 2;
    __bf16* Xv  = (__bf16*)(ws + off); off += XN * 2;
    __bf16* bWq = (__bf16*)(ws + off); off += WN * 2;
    __bf16* bWk = (__bf16*)(ws + off); off += WN * 2;
    __bf16* bWv = (__bf16*)(ws + off); off += WN * 2;
    __bf16* bWo = (__bf16*)(ws + off); off += WN * 2;
    __bf16* Qh  = (__bf16*)(ws + off); off += XN * 2;
    __bf16* Kh  = (__bf16*)(ws + off); off += XN * 2;
    __bf16* Vh  = (__bf16*)(ws + off); off += XN * 2;
    __bf16* Oh  = (__bf16*)(ws + off); off += XN * 2;
    unsigned* Mb = (unsigned*)(ws + off); off += (size_t)B_ * S_ * SW_ * 4;

    // f32 -> bf16 + mask bit-pack
    cvt_f32_bf16<<<1024, 256, 0, stream>>>(q,  Xq,  (int)XN);
    cvt_f32_bf16<<<1024, 256, 0, stream>>>(k,  Xk,  (int)XN);
    cvt_f32_bf16<<<1024, 256, 0, stream>>>(v,  Xv,  (int)XN);
    cvt_f32_bf16<<<512,  256, 0, stream>>>(Wq, bWq, (int)WN);
    cvt_f32_bf16<<<512,  256, 0, stream>>>(Wk, bWk, (int)WN);
    cvt_f32_bf16<<<512,  256, 0, stream>>>(Wv, bWv, (int)WN);
    cvt_f32_bf16<<<512,  256, 0, stream>>>(Wo, bWo, (int)WN);
    pack_mask<<<1024, 256, 0, stream>>>(mask, Mb, B_ * S_ * SW_);

    // QKV projections -> split-head bf16
    dim3 ggrid(D_ / 128, M_ / 128);
    gemm_bf16<0><<<ggrid, 256, 0, stream>>>(Xq, bWq, bq, Qh);
    gemm_bf16<0><<<ggrid, 256, 0, stream>>>(Xk, bWk, bk, Kh);
    gemm_bf16<0><<<ggrid, 256, 0, stream>>>(Xv, bWv, bv, Vh);

    // flash attention
    dim3 agrid(S_ / 64, H_, B_);
    attn<<<agrid, 128, 0, stream>>>(Qh, Kh, Vh, Mb, Oh);

    // output projection -> f32 d_out
    gemm_bf16<1><<<ggrid, 256, 0, stream>>>(Oh, bWo, bo, d_out);
}
